// Qwen2DyMemDecoderLayer_90924457656769
// MI455X (gfx1250) — matrix operand order repaired and checked by reading, not compiled, not run on hardware
//
#include <hip/hip_runtime.h>
#include <hip/hip_bf16.h>

// ---------------- constants (match reference) ----------------
#define T_      2048
#define H_      2048
#define NH_     16
#define KVH_    4
#define HD_     128
#define WIN_    512
#define SINK_   64
#define S_      64
#define NG_     4
#define MLPI_   5504
#define CK_     4
#define INTER_  2048
#define CONVD_  2560              // INTER + 2*NG*S
#define NPROJ_  4624              // INTER + CONVD + NH (real)
#define NPROJP_ 4864              // padded to multiple of 256 for GEMM
#define LMEM_   1472              // (T - WIN) - SINK
#define QKVN_   3072              // 2048 q + 512 k + 512 v
#define GUN_    11008             // gate + up fused

typedef _Float16 h4  __attribute__((ext_vector_type(4)));
typedef _Float16 h8  __attribute__((ext_vector_type(8)));
typedef __attribute__((ext_vector_type(16))) _Float16 v16h;
typedef __attribute__((ext_vector_type(8)))  float    v8f;

__device__ __forceinline__ v16h mkv(h8 lo, h8 hi) {
  return __builtin_shufflevector(lo, hi, 0,1,2,3,4,5,6,7,8,9,10,11,12,13,14,15);
}
__device__ __forceinline__ float silu_f(float x) { return x / (1.0f + __expf(-x)); }

__device__ __forceinline__ float block_reduce_sum_256(float v) {
  #pragma unroll
  for (int o = 16; o > 0; o >>= 1) v += __shfl_xor(v, o, 32);
  __shared__ float red[8];
  int w = threadIdx.x >> 5;
  if ((threadIdx.x & 31) == 0) red[w] = v;
  __syncthreads();
  float t = 0.f;
  #pragma unroll
  for (int i = 0; i < 8; ++i) t += red[i];
  __syncthreads();
  return t;
}

// ---------------- WMMA GEMM: C[M,N] = A[M,K] * W[N,K]^T (+bias) ----------------
// Requirements: N % 256 == 0, K % 64 == 0; A rows allocated to ceil(M/64)*64.
// Per wave: 32 rows x 64 cols (8 WMMAs per k-step of 32); double-buffered loads.
__global__ __launch_bounds__(256)
void gemm_wmma_f16(const _Float16* __restrict__ A, const _Float16* __restrict__ W,
                   const float* __restrict__ bias, float* __restrict__ C,
                   int M, int N, int K) {
  const int wave = threadIdx.x >> 5;
  const int lane = threadIdx.x & 31;
  const int wm = wave & 1, wn = wave >> 1;
  const int m0 = blockIdx.y * 64 + wm * 32;
  const int n0 = blockIdx.x * 256 + wn * 64;
  const int r  = lane & 15;
  const int hi = lane >> 4;

  const _Float16* a0 = A + (size_t)(m0 + r) * K;
  const _Float16* a1 = A + (size_t)(m0 + 16 + r) * K;
  const _Float16* w0 = W + (size_t)(n0 + r) * K;
  const _Float16* w1 = W + (size_t)(n0 + 16 + r) * K;
  const _Float16* w2 = W + (size_t)(n0 + 32 + r) * K;
  const _Float16* w3 = W + (size_t)(n0 + 48 + r) * K;

  v8f acc[8] = {};   // acc[s*2 + mt]
  h8 sa[2][4];
  h8 sb[2][8];

  auto LOAD = [&](int buf, int k) {
    sa[buf][0] = *(const h8*)(a0 + k + hi * 16);
    sa[buf][1] = *(const h8*)(a0 + k + hi * 16 + 8);
    sa[buf][2] = *(const h8*)(a1 + k + hi * 16);
    sa[buf][3] = *(const h8*)(a1 + k + hi * 16 + 8);
    sb[buf][0] = *(const h8*)(w0 + k + hi * 16);
    sb[buf][1] = *(const h8*)(w0 + k + hi * 16 + 8);
    sb[buf][2] = *(const h8*)(w1 + k + hi * 16);
    sb[buf][3] = *(const h8*)(w1 + k + hi * 16 + 8);
    sb[buf][4] = *(const h8*)(w2 + k + hi * 16);
    sb[buf][5] = *(const h8*)(w2 + k + hi * 16 + 8);
    sb[buf][6] = *(const h8*)(w3 + k + hi * 16);
    sb[buf][7] = *(const h8*)(w3 + k + hi * 16 + 8);
  };
  auto MMA = [&](int buf) {
    v16h va0 = mkv(sa[buf][0], sa[buf][1]);
    v16h va1 = mkv(sa[buf][2], sa[buf][3]);
    #pragma unroll
    for (int s = 0; s < 4; ++s) {
      v16h vb = mkv(sb[buf][s * 2], sb[buf][s * 2 + 1]);
      acc[s*2]   = __builtin_amdgcn_wmma_f32_16x16x32_f16(false, va0, false, vb, (short)0, acc[s*2],   false, false);
      acc[s*2+1] = __builtin_amdgcn_wmma_f32_16x16x32_f16(false, va1, false, vb, (short)0, acc[s*2+1], false, false);
    }
  };

  LOAD(0, 0);
  int k0 = 0;
  for (; k0 + 64 < K; k0 += 64) {
    LOAD(1, k0 + 32);
    MMA(0);
    LOAD(0, k0 + 64);
    MMA(1);
  }
  LOAD(1, k0 + 32);
  MMA(0);
  MMA(1);

  #pragma unroll
  for (int s = 0; s < 4; ++s) {
    int col = n0 + s * 16 + r;
    float bv = bias ? bias[col] : 0.0f;
    #pragma unroll
    for (int mt = 0; mt < 2; ++mt) {
      #pragma unroll
      for (int j = 0; j < 8; ++j) {
        int row = m0 + mt * 16 + j + hi * 8;
        if (row < M) C[(size_t)row * N + col] = acc[s*2+mt][j] + bv;
      }
    }
  }
}

// ---------------- WMMA flash-attention ----------------
// One wave per (head, 16-query tile). Key blocks of 32. Sliding window + sink.
__global__ __launch_bounds__(32)
void fa_wmma_kernel(const _Float16* __restrict__ qkvh, float* __restrict__ o) {
  const int bid = blockIdx.x;
  const int hh = bid & (NH_ - 1);
  const int r0 = (bid >> 4) * 16;
  const int lane = threadIdx.x;
  const int c16 = lane & 15;
  const int hi = lane >> 4;
  const int kvh = hh >> 2;
  const float scale = 0.08838834764831845f;   // 1/sqrt(128)

  __shared__ __align__(16) _Float16 Pl[16 * 40];
  __shared__ __align__(16) _Float16 Vt[128 * 40];

  // Q fragments (A-layout), 4 chunks of k=32 over head-dim 128
  v16h qf[4];
  {
    const _Float16* qrow = qkvh + (size_t)(r0 + c16) * QKVN_ + hh * HD_;
    #pragma unroll
    for (int ch = 0; ch < 4; ++ch) {
      h8 lo = *(const h8*)(qrow + ch * 32 + hi * 16);
      h8 hv = *(const h8*)(qrow + ch * 32 + hi * 16 + 8);
      qf[ch] = mkv(lo, hv);
    }
  }
  v8f O[8] = {};
  float mrow[8], lrow[8];
  #pragma unroll
  for (int j = 0; j < 8; ++j) { mrow[j] = -3.0e38f; lrow[j] = 0.f; }

  const int rmax = r0 + 15;

  auto process = [&](int kb) {
    // scores: S(16x32) = Q x K^T  (two 16x16 subtiles, 4 chained WMMAs each)
    v8f s0 = {}, s1 = {};
    int kr0 = kb + c16;       if (kr0 > T_) kr0 = T_;   // clamp to valid rows
    int kr1 = kb + 16 + c16;  if (kr1 > T_) kr1 = T_;
    const _Float16* kp0 = qkvh + (size_t)kr0 * QKVN_ + INTER_ + kvh * HD_;
    const _Float16* kp1 = qkvh + (size_t)kr1 * QKVN_ + INTER_ + kvh * HD_;
    #pragma unroll
    for (int ch = 0; ch < 4; ++ch) {
      h8 bl0 = *(const h8*)(kp0 + ch * 32 + hi * 16);
      h8 bh0 = *(const h8*)(kp0 + ch * 32 + hi * 16 + 8);
      v16h b0 = mkv(bl0, bh0);
      s0 = __builtin_amdgcn_wmma_f32_16x16x32_f16(false, qf[ch], false, b0, (short)0, s0, false, false);
      h8 bl1 = *(const h8*)(kp1 + ch * 32 + hi * 16);
      h8 bh1 = *(const h8*)(kp1 + ch * 32 + hi * 16 + 8);
      v16h b1 = mkv(bl1, bh1);
      s1 = __builtin_amdgcn_wmma_f32_16x16x32_f16(false, qf[ch], false, b1, (short)0, s1, false, false);
    }
    // stage V block (32 keys x 128 dims) transposed into LDS
    #pragma unroll 4
    for (int kk = 0; kk < 32; ++kk) {
      int vr = kb + kk; if (vr > T_) vr = T_;
      const _Float16* vp = qkvh + (size_t)vr * QKVN_ + INTER_ + 512 + kvh * HD_ + lane * 4;
      h4 vv = *(const h4*)vp;
      #pragma unroll
      for (int t = 0; t < 4; ++t) Vt[(lane * 4 + t) * 40 + kk] = vv[t];
    }
    // mask + online softmax (row = r0 + j + 8*hi, col = kb [+16] + c16)
    #pragma unroll
    for (int j = 0; j < 8; ++j) {
      int row = r0 + j + hi * 8;
      int col0 = kb + c16;
      int col1 = col0 + 16;
      bool a0 = (col0 <= row) && ((col0 <= SINK_) || (col0 >= row - WIN_));
      bool a1 = (col1 <= row) && ((col1 <= SINK_) || (col1 >= row - WIN_));
      float v0 = a0 ? s0[j] * scale + (col0 == SINK_ ? 0.359375f : 0.f) : -3.0e38f;
      float v1 = a1 ? s1[j] * scale : -3.0e38f;           // col1 is never 64
      float bm = fmaxf(v0, v1);
      bm = fmaxf(bm, __shfl_xor(bm, 1, 32));
      bm = fmaxf(bm, __shfl_xor(bm, 2, 32));
      bm = fmaxf(bm, __shfl_xor(bm, 4, 32));
      bm = fmaxf(bm, __shfl_xor(bm, 8, 32));
      float mn = fmaxf(mrow[j], bm);
      float f  = __expf(mrow[j] - mn);
      float p0 = __expf(v0 - mn);
      float p1 = __expf(v1 - mn);
      float rs = p0 + p1;
      rs += __shfl_xor(rs, 1, 32);
      rs += __shfl_xor(rs, 2, 32);
      rs += __shfl_xor(rs, 4, 32);
      rs += __shfl_xor(rs, 8, 32);
      lrow[j] = lrow[j] * f + rs;
      mrow[j] = mn;
      #pragma unroll
      for (int ch = 0; ch < 8; ++ch) O[ch][j] *= f;
      int prow = j + hi * 8;
      Pl[prow * 40 + c16]      = (_Float16)p0;
      Pl[prow * 40 + 16 + c16] = (_Float16)p1;
    }
    __syncthreads();
    // O(16x128) += P(16x32) x V(32x128): 8 WMMAs
    const _Float16* pr = Pl + c16 * 40;
    h8 plo = *(const h8*)(pr + hi * 16);
    h8 phi = *(const h8*)(pr + hi * 16 + 8);
    v16h pa = mkv(plo, phi);
    #pragma unroll
    for (int ch = 0; ch < 8; ++ch) {
      const _Float16* vr = Vt + (size_t)(ch * 16 + c16) * 40 + hi * 16;
      h8 vlo = *(const h8*)vr;
      h8 vhi = *(const h8*)(vr + 8);
      v16h vb = mkv(vlo, vhi);
      O[ch] = __builtin_amdgcn_wmma_f32_16x16x32_f16(false, pa, false, vb, (short)0, O[ch], false, false);
    }
    __syncthreads();
  };

  int e1 = rmax < 95 ? rmax : 95;          // blocks 0,32,64 (cols 0..95: sink region)
  for (int kb = 0; kb <= e1; kb += 32) process(kb);
  int start2 = (r0 - WIN_) & ~31;          // window region, disjoint from blocks < 96
  if (start2 < 96) start2 = 96;
  for (int kb = start2; kb <= rmax; kb += 32) process(kb);

  #pragma unroll
  for (int j = 0; j < 8; ++j) {
    float inv = 1.0f / lrow[j];
    int row = r0 + j + hi * 8;
    float* orow = o + (size_t)row * INTER_ + hh * HD_;
    #pragma unroll
    for (int ch = 0; ch < 8; ++ch)
      orow[ch * 16 + c16] = O[ch][j] * inv;
  }
}

// ---------------- elementwise / helper kernels ----------------
__global__ void cvt_f16_kernel(const float* __restrict__ s, _Float16* __restrict__ d, size_t n) {
  size_t i = blockIdx.x * (size_t)blockDim.x + threadIdx.x;
  if (i < n) d[i] = (_Float16)s[i];
}
__global__ void copy_f32_kernel(const float* __restrict__ s, float* __restrict__ d, size_t n) {
  size_t i = blockIdx.x * (size_t)blockDim.x + threadIdx.x;
  if (i < n) d[i] = s[i];
}
__global__ void add_f32_kernel(const float* __restrict__ a, const float* __restrict__ b,
                               float* __restrict__ o, size_t n) {
  size_t i = blockIdx.x * (size_t)blockDim.x + threadIdx.x;
  if (i < n) o[i] = a[i] + b[i];
}

__global__ __launch_bounds__(256)
void rmsnorm_f16_kernel(const float* __restrict__ x, const float* __restrict__ w,
                        _Float16* __restrict__ out, int ncols, float eps) {
  int row = blockIdx.x;
  const float* xr = x + (size_t)row * ncols;
  float ss = 0.f;
  for (int i = threadIdx.x; i < ncols; i += 256) { float v = xr[i]; ss += v * v; }
  float tot = block_reduce_sum_256(ss);
  float rs = rsqrtf(tot / (float)ncols + eps);
  for (int i = threadIdx.x; i < ncols; i += 256)
    out[(size_t)row * ncols + i] = (_Float16)(xr[i] * rs * w[i]);
}

// conv1d (causal, depthwise, CK=4) + bias + silu on xBC slice of proj (stride NPROJP_)
__global__ void conv_silu_kernel(const float* __restrict__ proj, const float* __restrict__ cw,
                                 const float* __restrict__ cb, float* __restrict__ xbc) {
  size_t idx = blockIdx.x * (size_t)blockDim.x + threadIdx.x;
  if (idx >= (size_t)LMEM_ * CONVD_) return;
  int c = (int)(idx % CONVD_);
  int l = (int)(idx / CONVD_);
  float acc = cb[c];
  #pragma unroll
  for (int j = 0; j < CK_; ++j) {
    int li = l - (CK_ - 1) + j;
    if (li >= 0) acc += proj[(size_t)li * NPROJP_ + INTER_ + c] * cw[c * CK_ + j];
  }
  xbc[idx] = silu_f(acc);
}

__global__ void dt_kernel(const float* __restrict__ proj, const float* __restrict__ dt_bias,
                          float* __restrict__ dtb) {
  int idx = blockIdx.x * blockDim.x + threadIdx.x;
  if (idx >= LMEM_ * NH_) return;
  int hh = idx & (NH_ - 1);
  int l  = idx >> 4;
  float x = proj[(size_t)l * NPROJP_ + INTER_ + CONVD_ + hh] + dt_bias[hh];
  dtb[idx] = (x > 20.f) ? x : log1pf(__expf(x));
}

// selective-state scan: 64 blocks x 256 thr; thread owns (h, d) with 8 of 64 states.
// B/C read directly from global as float4 (L2-resident) -> no barriers, pipelineable.
__global__ __launch_bounds__(256)
void scan_kernel(const float* __restrict__ xbc, const float* __restrict__ dtb,
                 const float* __restrict__ A_log, const float* __restrict__ Dv,
                 float* __restrict__ ybuf) {
  int tid = threadIdx.x;
  int hh = blockIdx.x >> 2;                 // 4 blocks per head
  int d  = (blockIdx.x & 3) * 32 + (tid >> 3);
  int chunk = tid & 7;
  int s0 = chunk * 8;
  int g  = hh >> 2;
  float Ah = -__expf(A_log[hh]);
  float Dh = Dv[hh];
  float st[8];
  #pragma unroll
  for (int s = 0; s < 8; ++s) st[s] = 0.f;
  for (int l = 0; l < LMEM_; ++l) {
    const float* xr = xbc + (size_t)l * CONVD_;
    float dtv = dtb[l * NH_ + hh];
    float x = xr[hh * HD_ + d];
    float4 b0 = *(const float4*)(xr + INTER_ + g * S_ + s0);
    float4 b1 = *(const float4*)(xr + INTER_ + g * S_ + s0 + 4);
    float4 c0 = *(const float4*)(xr + INTER_ + NG_ * S_ + g * S_ + s0);
    float4 c1 = *(const float4*)(xr + INTER_ + NG_ * S_ + g * S_ + s0 + 4);
    float decay = __expf(dtv * Ah);
    float cx = dtv * x;
    float y;
    st[0] = decay * st[0] + cx * b0.x; y  = st[0] * c0.x;
    st[1] = decay * st[1] + cx * b0.y; y += st[1] * c0.y;
    st[2] = decay * st[2] + cx * b0.z; y += st[2] * c0.z;
    st[3] = decay * st[3] + cx * b0.w; y += st[3] * c0.w;
    st[4] = decay * st[4] + cx * b1.x; y += st[4] * c1.x;
    st[5] = decay * st[5] + cx * b1.y; y += st[5] * c1.y;
    st[6] = decay * st[6] + cx * b1.z; y += st[6] * c1.z;
    st[7] = decay * st[7] + cx * b1.w; y += st[7] * c1.w;
    y += __shfl_xor(y, 1, 32);
    y += __shfl_xor(y, 2, 32);
    y += __shfl_xor(y, 4, 32);
    if (chunk == 0) ybuf[(size_t)l * INTER_ + hh * HD_ + d] = y + Dh * x;
  }
}

// g = y*silu(z); rmsnorm(g)*mnorm_w -> f16
__global__ __launch_bounds__(256)
void gatednorm_kernel(const float* __restrict__ ybuf, const float* __restrict__ proj,
                      const float* __restrict__ w, _Float16* __restrict__ out) {
  int l = blockIdx.x;
  const float* yr = ybuf + (size_t)l * INTER_;
  const float* zr = proj + (size_t)l * NPROJP_;
  float ss = 0.f;
  for (int i = threadIdx.x; i < INTER_; i += 256) {
    float gg = yr[i] * silu_f(zr[i]);
    ss += gg * gg;
  }
  float tot = block_reduce_sum_256(ss);
  float rs = rsqrtf(tot / (float)INTER_ + 1e-5f);
  for (int i = threadIdx.x; i < INTER_; i += 256) {
    float gg = yr[i] * silu_f(zr[i]);
    out[(size_t)l * INTER_ + i] = (_Float16)(gg * rs * w[i]);
  }
}

__global__ void memact_kernel(const float* __restrict__ mem_out, _Float16* __restrict__ act_h) {
  int i = blockIdx.x * blockDim.x + threadIdx.x;
  if (i < H_) act_h[(size_t)T_ * H_ + i] = (_Float16)mem_out[(size_t)(LMEM_ - 1) * INTER_ + i];
}

// RoPE in-place on fused qkv rows t<T (q: 16 heads, k: 4 heads)
__global__ void rope_kernel(float* __restrict__ qkv, const float* __restrict__ cosb,
                            const float* __restrict__ sinb) {
  size_t idx = blockIdx.x * (size_t)blockDim.x + threadIdx.x;
  if (idx >= (size_t)T_ * 20 * 64) return;
  int d = (int)(idx & 63);
  int head = (int)((idx >> 6) % 20);
  int t = (int)(idx / (64 * 20));
  int base = head < 16 ? head * HD_ : INTER_ + (head - 16) * HD_;
  float* p = qkv + (size_t)t * QKVN_ + base;
  float c = cosb[t * HD_ + d];
  float s = sinb[t * HD_ + d];
  float x0 = p[d], x1 = p[d + 64];
  p[d]      = x0 * c - x1 * s;
  p[d + 64] = x1 * c + x0 * s;
}

__global__ void sinkset_kernel(float* __restrict__ qkv) {
  int i = blockIdx.x * blockDim.x + threadIdx.x;
  if (i < QKVN_) qkv[(size_t)SINK_ * QKVN_ + i] = qkv[(size_t)T_ * QKVN_ + i];
}

// merged = concat(o[:SINK], mem_out, o[T-WIN:]) -> f16
__global__ void merged_kernel(const float* __restrict__ o, const float* __restrict__ mem_out,
                              _Float16* __restrict__ merged) {
  size_t idx = blockIdx.x * (size_t)blockDim.x + threadIdx.x;
  if (idx >= (size_t)T_ * INTER_) return;
  int t = (int)(idx / INTER_);
  int i = (int)(idx % INTER_);
  float v;
  if (t < SINK_)          v = o[idx];
  else if (t < T_ - WIN_) v = mem_out[(size_t)(t - SINK_) * INTER_ + i];
  else                    v = o[idx];
  merged[idx] = (_Float16)v;
}

__global__ void silumul_kernel(const float* __restrict__ gu, _Float16* __restrict__ out) {
  size_t idx = blockIdx.x * (size_t)blockDim.x + threadIdx.x;
  if (idx >= (size_t)T_ * MLPI_) return;
  int t = (int)(idx / MLPI_);
  int i = (int)(idx % MLPI_);
  float g = gu[(size_t)t * GUN_ + i];
  float u = gu[(size_t)t * GUN_ + MLPI_ + i];
  out[idx] = (_Float16)(silu_f(g) * u);
}

// ---------------- host launcher ----------------
static inline size_t cdiv_(size_t a, size_t b) { return (a + b - 1) / b; }

extern "C" void kernel_launch(void* const* d_in, const int* in_sizes, int n_in,
                              void* d_out, int out_size, void* d_ws, size_t ws_size,
                              hipStream_t stream) {
  (void)in_sizes; (void)n_in; (void)out_size; (void)ws_size;
  const float* hidden    = (const float*)d_in[0];
  const float* cosb      = (const float*)d_in[1];
  const float* sinb      = (const float*)d_in[2];
  const float* ln1_w     = (const float*)d_in[3];
  const float* q_w       = (const float*)d_in[4];
  const float* q_b       = (const float*)d_in[5];
  const float* k_w       = (const float*)d_in[6];
  const float* k_b       = (const float*)d_in[7];
  const float* v_w       = (const float*)d_in[8];
  const float* v_b       = (const float*)d_in[9];
  const float* o_w       = (const float*)d_in[10];
  const float* in_proj_w = (const float*)d_in[11];
  const float* conv_w    = (const float*)d_in[12];
  const float* conv_b    = (const float*)d_in[13];
  const float* dt_bias   = (const float*)d_in[14];
  const float* A_log     = (const float*)d_in[15];
  const float* D_in      = (const float*)d_in[16];
  const float* mnorm_w   = (const float*)d_in[17];
  const float* out_proj_w= (const float*)d_in[18];
  const float* ln2_w     = (const float*)d_in[19];
  const float* gate_w    = (const float*)d_in[20];
  const float* up_w      = (const float*)d_in[21];
  const float* down_w    = (const float*)d_in[22];
  float* out = (float*)d_out;

  // --- workspace layout (bump allocator, 256B aligned) ---
  char* base = (char*)d_ws;
  size_t off = 0;
  auto alloc = [&](size_t bytes) -> char* {
    char* p = base + off;
    off = (off + bytes + 255) & ~(size_t)255;
    return p;
  };
  // f16 weights
  _Float16* w_qkv  = (_Float16*)alloc((size_t)QKVN_ * H_ * 2);
  _Float16* w_inpj = (_Float16*)alloc((size_t)NPROJP_ * H_ * 2);  // padded rows = garbage
  _Float16* w_o    = (_Float16*)alloc((size_t)H_ * INTER_ * 2);
  _Float16* w_outp = (_Float16*)alloc((size_t)H_ * INTER_ * 2);
  _Float16* w_gu   = (_Float16*)alloc((size_t)GUN_ * H_ * 2);
  _Float16* w_down = (_Float16*)alloc((size_t)H_ * MLPI_ * 2);
  float*    qkv_b  = (float*)alloc((size_t)QKVN_ * 4);
  // activations
  _Float16* act_h   = (_Float16*)alloc((size_t)2112 * H_ * 2);   // rows 0..T-1 pre; row T = mem_act
  float*    mem_out = (float*)alloc((size_t)LMEM_ * INTER_ * 4);
  float*    dtb     = (float*)alloc((size_t)LMEM_ * NH_ * 4);
  _Float16* g_h     = (_Float16*)alloc((size_t)LMEM_ * INTER_ * 2);
  float*    qkv_f   = (float*)alloc((size_t)2112 * QKVN_ * 4);
  _Float16* qkv_h   = (_Float16*)alloc((size_t)2112 * QKVN_ * 2);
  float*    o_f     = (float*)alloc((size_t)T_ * INTER_ * 4);
  _Float16* merged  = (_Float16*)alloc((size_t)T_ * INTER_ * 2);
  float*    attn_f  = (float*)alloc((size_t)T_ * H_ * 4);
  float*    h_f     = (float*)alloc((size_t)T_ * H_ * 4);
  _Float16* m_h     = (_Float16*)alloc((size_t)T_ * H_ * 2);
  _Float16* mact_h  = (_Float16*)alloc((size_t)T_ * MLPI_ * 2);
  float*    mlp_f   = (float*)alloc((size_t)T_ * H_ * 4);
  // aliased scratch region: stage-1 (proj,xbc,y) vs stage-3 (gate|up)
  size_t szProj = ((size_t)LMEM_ * NPROJP_ * 4 + 255) & ~(size_t)255;
  size_t szXbc  = ((size_t)LMEM_ * CONVD_ * 4 + 255) & ~(size_t)255;
  size_t szY    = (size_t)LMEM_ * INTER_ * 4;
  size_t szGu   = (size_t)T_ * GUN_ * 4;
  size_t regSz  = szProj + szXbc + szY; if (szGu > regSz) regSz = szGu;
  char* region = alloc(regSz);
  float* proj = (float*)region;
  float* xbc  = (float*)(region + szProj);
  float* ybuf = (float*)(region + szProj + szXbc);
  float* gu   = (float*)region;

  const int EW = 256;
  auto blocks = [&](size_t n) { return (unsigned)cdiv_(n, EW); };
  auto cvt = [&](const float* s, _Float16* d, size_t n) {
    cvt_f16_kernel<<<blocks(n), EW, 0, stream>>>(s, d, n);
  };
  auto launch_gemm = [&](const _Float16* A, const _Float16* W, const float* bias,
                         float* C, int M, int N, int K) {
    int rowsAlloc = ((M + 63) / 64) * 64;
    dim3 grid((unsigned)(N / 256), (unsigned)(rowsAlloc / 64));
    gemm_wmma_f16<<<grid, 256, 0, stream>>>(A, W, bias, C, M, N, K);
  };

  // 1) weight conversions (f32 -> f16), fused layouts
  cvt(q_w,       w_qkv,                             (size_t)H_ * H_);
  cvt(k_w,       w_qkv + (size_t)H_ * H_,           (size_t)512 * H_);
  cvt(v_w,       w_qkv + (size_t)(H_ + 512) * H_,   (size_t)512 * H_);
  cvt(in_proj_w, w_inpj,                            (size_t)NPROJ_ * H_);
  cvt(o_w,       w_o,                               (size_t)H_ * INTER_);
  cvt(out_proj_w,w_outp,                            (size_t)H_ * INTER_);
  cvt(gate_w,    w_gu,                              (size_t)MLPI_ * H_);
  cvt(up_w,      w_gu + (size_t)MLPI_ * H_,         (size_t)MLPI_ * H_);
  cvt(down_w,    w_down,                            (size_t)H_ * MLPI_);
  copy_f32_kernel<<<blocks(H_), EW, 0, stream>>>(q_b, qkv_b, (size_t)H_);
  copy_f32_kernel<<<blocks(512), EW, 0, stream>>>(k_b, qkv_b + H_, (size_t)512);
  copy_f32_kernel<<<blocks(512), EW, 0, stream>>>(v_b, qkv_b + H_ + 512, (size_t)512);

  // 2) pre = rmsnorm(hidden, ln1_w) -> f16
  rmsnorm_f16_kernel<<<T_, 256, 0, stream>>>(hidden, ln1_w, act_h, H_, 1e-6f);

  // 3) mamba branch on rows [SINK, SINK+LMEM)
  launch_gemm(act_h + (size_t)SINK_ * H_, w_inpj, nullptr, proj, LMEM_, NPROJP_, H_);
  conv_silu_kernel<<<blocks((size_t)LMEM_ * CONVD_), EW, 0, stream>>>(proj, conv_w, conv_b, xbc);
  dt_kernel<<<blocks((size_t)LMEM_ * NH_), EW, 0, stream>>>(proj, dt_bias, dtb);
  scan_kernel<<<NH_ * 4, 256, 0, stream>>>(xbc, dtb, A_log, D_in, ybuf);
  gatednorm_kernel<<<LMEM_, 256, 0, stream>>>(ybuf, proj, mnorm_w, g_h);
  launch_gemm(g_h, w_outp, nullptr, mem_out, LMEM_, H_, INTER_);
  memact_kernel<<<blocks(H_), EW, 0, stream>>>(mem_out, act_h);

  // 4) fused QKV over T+1 rows (last row = mem_act), RoPE, sink replacement, f16 copy
  launch_gemm(act_h, w_qkv, qkv_b, qkv_f, T_ + 1, QKVN_, H_);
  rope_kernel<<<blocks((size_t)T_ * 20 * 64), EW, 0, stream>>>(qkv_f, cosb, sinb);
  sinkset_kernel<<<blocks(QKVN_), EW, 0, stream>>>(qkv_f);
  cvt(qkv_f, qkv_h, (size_t)(T_ + 1) * QKVN_);

  // 5) WMMA flash attention: one wave per (head, 16-query tile)
  fa_wmma_kernel<<<(T_ / 16) * NH_, 32, 0, stream>>>(qkv_h, o_f);

  // 6) merge, o-projection, residual, rmsnorm
  merged_kernel<<<blocks((size_t)T_ * INTER_), EW, 0, stream>>>(o_f, mem_out, merged);
  launch_gemm(merged, w_o, nullptr, attn_f, T_, H_, INTER_);
  add_f32_kernel<<<blocks((size_t)T_ * H_), EW, 0, stream>>>(hidden, attn_f, h_f, (size_t)T_ * H_);
  rmsnorm_f16_kernel<<<T_, 256, 0, stream>>>(h_f, ln2_w, m_h, H_, 1e-6f);

  // 7) MLP: fused gate+up GEMM, silu*mul, down GEMM, final residual
  launch_gemm(m_h, w_gu, nullptr, gu, T_, GUN_, H_);
  silumul_kernel<<<blocks((size_t)T_ * MLPI_), EW, 0, stream>>>(gu, mact_h);
  launch_gemm(mact_h, w_down, nullptr, mlp_f, T_, H_, MLPI_);
  add_f32_kernel<<<blocks((size_t)T_ * H_), EW, 0, stream>>>(h_f, mlp_f, out, (size_t)T_ * H_);
}
